// PriorConditionedSelectiveStateSpace_25984552140981
// MI455X (gfx1250) — compile-verified, matrix operand order checked by reading, not run on hardware
//
#include <hip/hip_runtime.h>
#include <hip/hip_bf16.h>

typedef __attribute__((ext_vector_type(16))) _Float16 v16h;
typedef __attribute__((ext_vector_type(8)))  _Float16 v8h;
typedef __attribute__((ext_vector_type(8)))  float    v8f;

#define HID 128
#define CIN 64
#define SPX 65536   // 256*256
#define NB  4
#define FR  24      // f16 stride per lane in fragment LDS buffers (48B, bank-tiled)
#define FRW 12      // uint32 stride per lane

// K index inside a 16-bit 16x32 WMMA A/B fragment (ISA 7.12.2):
//  lanes 0-15 : e=0..7 -> K=e    ; e=8..15 -> K=16+(e-8)
//  lanes 16-31: e=0..7 -> K=8+e  ; e=8..15 -> K=24+(e-8)
__device__ __forceinline__ int frag_k(int e, int half) {
    return (e & 7) + (half << 3) + ((e >> 3) << 4);
}
// inverse: row kk (0..31) at column n -> (lane, e) in fragment
__device__ __forceinline__ void frag_pos(int kk, int n, int& lane, int& e) {
    const int half = (kk >> 3) & 1;
    e = (kk & 7) + ((kk >> 4) << 3);
    lane = (half << 4) + n;
}

__device__ __forceinline__ uint32_t pack2(float f0, float f1) {
    union { _Float16 h; unsigned short u; } a, b;
    a.h = (_Float16)f0; b.h = (_Float16)f1;
    return (uint32_t)a.u | ((uint32_t)b.u << 16);
}

__device__ __forceinline__ v8f wmma16(v16h a, v16h b, v8f c) {
    return __builtin_amdgcn_wmma_f32_16x16x32_f16(false, a, false, b, (short)0, c,
                                                  false, false);
}

__device__ __forceinline__ float softplusf(float v) {
    return v > 20.0f ? v : log1pf(expf(v));
}

// -------------------------------------------------- pack weights -> f16 fragments
__global__ void __launch_bounds__(256)
k_pack(const float* __restrict__ W, _Float16* __restrict__ Wp, int Kc, int K, int total)
{
    const int i = blockIdx.x * 256 + threadIdx.x;
    if (i >= total) return;
    const int e    = i & 15;
    const int lane = (i >> 4) & 31;
    const int seg  = i >> 9;
    const int kc   = seg % Kc;
    const int mt   = seg / Kc;
    const int row  = mt * 16 + (lane & 15);
    const int k    = kc * 32 + frag_k(e, lane >> 4);
    Wp[i] = (_Float16)W[row * K + k];
}

// ---------------------------------------------------------------- prior upsample
__global__ void __launch_bounds__(256)
k_prior(const float* __restrict__ prior, float* __restrict__ pu)
{
    const int i  = blockIdx.x * 256 + threadIdx.x;   // exactly 4*65536 threads
    const int b  = i >> 16;
    const int s  = i & 65535;
    const int oy = s >> 8, ox = s & 255;
    const float sy = oy * (63.0f / 255.0f);
    const float sx = ox * (63.0f / 255.0f);
    const int y0 = (int)sy, x0 = (int)sx;
    const int y1 = min(y0 + 1, 63), x1 = min(x0 + 1, 63);
    const float wy = sy - (float)y0, wx = sx - (float)x0;
    const float* p = prior + (size_t)b * 4096;
    const float r0 = p[y0 * 64 + x0] * (1.0f - wx) + p[y0 * 64 + x1] * wx;
    const float r1 = p[y1 * 64 + x0] * (1.0f - wx) + p[y1 * 64 + x1] * wx;
    float v = r0 * (1.0f - wy) + r1 * wy;
    pu[i] = fminf(1.0f, fmaxf(-1.0f, v));
}

// ------------------------------------------- fused in/delta/B projections (WMMA)
// 64 pixels per block, 8 waves; fragment-ordered LDS; weights live in registers.
__global__ void __launch_bounds__(256)
k_proj(const float* __restrict__ x, const float* __restrict__ pu,
       const _Float16* __restrict__ Wp_in, const float* __restrict__ b_in,
       const _Float16* __restrict__ Wp_d, const float* __restrict__ b_delta,
       const _Float16* __restrict__ Wp_B, const float* __restrict__ b_B,
       const float* __restrict__ A_param,
       const float* __restrict__ p_lambda, const float* __restrict__ p_alpha,
       _Float16* __restrict__ xp_g, _Float16* __restrict__ Ab_g,
       _Float16* __restrict__ Bb_g)
{
    __shared__ uint32_t xsf[2 * 4 * 32 * FRW];   // x tile, fragment order
    __shared__ uint32_t xpf[4 * 4 * 32 * FRW];   // x_proj tile, fragment order

    const int tid  = threadIdx.x;
    const int wave = tid >> 5, lane = tid & 31;
    const int half = lane >> 4, n = lane & 15;
    const int pix0 = blockIdx.x << 6;            // 64 px, one row, one batch
    const int b    = pix0 >> 16;
    const int s0   = pix0 & 65535;
    const int mo   = wave << 4;

    // ---- weight fragments, loaded once (contiguous 32B per lane)
    const v16h* Wi = (const v16h*)Wp_in;
    const v16h* Wd = (const v16h*)Wp_d;
    const v16h* Wb = (const v16h*)Wp_B;
    v16h ai[2], ad[4], ab[4];
    #pragma unroll
    for (int kc = 0; kc < 2; ++kc) ai[kc] = Wi[(wave * 2 + kc) * 32 + lane];
    #pragma unroll
    for (int kc = 0; kc < 4; ++kc) {
        ad[kc] = Wd[(wave * 4 + kc) * 32 + lane];
        ab[kc] = Wb[(wave * 4 + kc) * 32 + lane];
    }

    // ---- stage x tile directly into fragment order (packed f16 pair stores)
    #pragma unroll
    for (int it = 0; it < 8; ++it) {
        const int j  = it * 256 + tid;           // 0..2047 channel-pairs x px
        const int px = j & 63;
        const int c  = (j >> 6) << 1;            // even channel
        const float f0 = x[((size_t)(b * CIN + c) << 16) + s0 + px];
        const float f1 = x[((size_t)(b * CIN + c + 1) << 16) + s0 + px];
        const int t = px >> 4;
        int laneC, e; frag_pos(c & 31, px & 15, laneC, e);
        xsf[(((c >> 5) * 4 + t) * 32 + laneC) * FRW + (e >> 1)] = pack2(f0, f1);
    }
    __syncthreads();

    // ---- GEMM1: x_proj = W_in(128x64) * x, 4 N-tiles
    #pragma unroll
    for (int t = 0; t < 4; ++t) {
        v8f acc = {};
        #pragma unroll
        for (int kc = 0; kc < 2; ++kc) {
            const v16h bf =
                *(const v16h*)((const _Float16*)xsf + ((kc * 4 + t) * 32 + lane) * FR);
            acc = wmma16(ai[kc], bf, acc);
        }
        const int s = s0 + t * 16 + n;
        #pragma unroll
        for (int r = 0; r < 8; r += 2) {
            const int o = mo + (half << 3) + r;
            const float v0 = acc[r] + b_in[o];
            const float v1 = acc[r + 1] + b_in[o + 1];
            xp_g[((size_t)(b * HID + o) << 16) + s]       = (_Float16)v0;
            xp_g[((size_t)(b * HID + o + 1) << 16) + s]   = (_Float16)v1;
            int laneC, e; frag_pos(o & 31, n, laneC, e);
            xpf[(((o >> 5) * 4 + t) * 32 + laneC) * FRW + (e >> 1)] = pack2(v0, v1);
        }
    }
    __syncthreads();

    // ---- per-lane invariants for the epilogue
    const float lam = p_lambda[0], alp = p_alpha[0];
    float bD[8], bB[8], aE[8];
    #pragma unroll
    for (int r = 0; r < 8; ++r) {
        const int o = mo + (half << 3) + r;
        bD[r] = b_delta[o];
        bB[r] = b_B[o];
        aE[r] = expf(A_param[o]);
    }

    // ---- GEMM2+3: W_delta * xp ; W_B * xp  (K=128), fused SSM epilogue
    #pragma unroll
    for (int t = 0; t < 4; ++t) {
        v8f accD = {}, accB = {};
        #pragma unroll
        for (int kc = 0; kc < 4; ++kc) {
            const v16h bf =
                *(const v16h*)((const _Float16*)xpf + ((kc * 4 + t) * 32 + lane) * FR);
            accD = wmma16(ad[kc], bf, accD);
            accB = wmma16(ab[kc], bf, accB);
        }
        const int s = s0 + t * 16 + n;
        const float pr = pu[((size_t)b << 16) + s];
        #pragma unroll
        for (int r = 0; r < 8; ++r) {
            const int o = mo + (half << 3) + r;
            const float delta = softplusf(accD[r] + bD[r] + lam * pr);
            const float abar  = expf(-delta * aE[r]);
            const float bbar  = delta * (accB[r] + bB[r]) * (1.0f + alp * pr);
            const size_t idx  = ((size_t)(b * HID + o) << 16) + s;
            Ab_g[idx] = (_Float16)abar;
            Bb_g[idx] = (_Float16)bbar;
        }
    }
}

// ------------------------------- scan along W: wave-parallel linear recurrence
__global__ void __launch_bounds__(256)
k_scan_w(const _Float16* __restrict__ xp, const _Float16* __restrict__ A,
         const _Float16* __restrict__ B, _Float16* __restrict__ S)
{
    const int gid  = blockIdx.x * 256 + threadIdx.x;
    const int row  = gid >> 5;                 // NB*HID*256 rows
    const int lane = threadIdx.x & 31;
    const size_t base = ((size_t)row << 8) + lane * 8;

    const v8h a8 = *(const v8h*)(A + base);
    const v8h b8 = *(const v8h*)(B + base);
    const v8h x8 = *(const v8h*)(xp + base);

    float aSeg = 1.0f, bSeg = 0.0f;
    #pragma unroll
    for (int j = 0; j < 8; ++j) {
        const float aj = (float)a8[j];
        bSeg = aj * bSeg + (float)b8[j] * (float)x8[j];
        aSeg *= aj;
    }
    float aInc = aSeg, bInc = bSeg;
    #pragma unroll
    for (int off = 1; off < 32; off <<= 1) {
        const float aUp = __shfl_up(aInc, off, 32);
        const float bUp = __shfl_up(bInc, off, 32);
        if (lane >= off) {
            bInc = bUp * aInc + bInc;
            aInc = aUp * aInc;
        }
    }
    float hIn = __shfl_up(bInc, 1, 32);
    if (lane == 0) hIn = 0.0f;

    v8h s8;
    float h = hIn;
    #pragma unroll
    for (int j = 0; j < 8; ++j) {
        h = (float)a8[j] * h + (float)b8[j] * (float)x8[j];
        s8[j] = (_Float16)h;
    }
    *(v8h*)(S + base) = s8;
}

// ----------------------------------------------- scan along H (coalesced), S +=
__global__ void __launch_bounds__(256)
k_scan_h(const _Float16* __restrict__ xp, const _Float16* __restrict__ A,
         const _Float16* __restrict__ B, _Float16* __restrict__ S)
{
    const int col = blockIdx.x * 256 + threadIdx.x;  // NB*HID*256 columns
    const int bc  = col >> 8;
    const int w   = col & 255;
    const size_t base = ((size_t)bc << 16) + w;
    float h = 0.0f;
    for (int hh = 0; hh < 256; ++hh) {
        const size_t i = base + ((size_t)hh << 8);
        if ((hh & 7) == 0 && hh < 248) {             // stride prefetch 8 rows ahead
            __builtin_prefetch(A + i + (8 << 8), 0, 0);
            __builtin_prefetch(B + i + (8 << 8), 0, 0);
            __builtin_prefetch(xp + i + (8 << 8), 0, 0);
        }
        h = (float)A[i] * h + (float)B[i] * (float)xp[i];
        S[i] = (_Float16)((float)S[i] + h);
    }
}

// ------------------------------------------------- out = x + gamma*(W_out*S + b)
__global__ void __launch_bounds__(128)
k_out(const float* __restrict__ x, const _Float16* __restrict__ S,
      const _Float16* __restrict__ Wp_o, const float* __restrict__ b_out,
      const float* __restrict__ p_gamma, float* __restrict__ out)
{
    __shared__ uint32_t Sf[4 * 4 * 32 * FRW];    // S tile, fragment order

    const int tid  = threadIdx.x;
    const int wave = tid >> 5, lane = tid & 31;
    const int half = lane >> 4, n = lane & 15;
    const int pix0 = blockIdx.x << 6;            // 64 px per block
    const int b    = pix0 >> 16;
    const int s0   = pix0 & 65535;
    const int mo   = wave << 4;                  // 4 waves cover 64 out channels

    const v16h* Wo = (const v16h*)Wp_o;
    v16h aw[4];
    #pragma unroll
    for (int kc = 0; kc < 4; ++kc) aw[kc] = Wo[(wave * 4 + kc) * 32 + lane];

    // stage S tile (128ch x 64px) into fragment order; coalesced u16 pair loads
    const unsigned short* S16 = (const unsigned short*)S;
    #pragma unroll
    for (int it = 0; it < 32; ++it) {
        const int j  = it * 128 + tid;           // 0..4095 channel-pairs x px
        const int px = j & 63;
        const int c  = (j >> 6) << 1;
        const uint32_t u0 = S16[((size_t)(b * HID + c) << 16) + s0 + px];
        const uint32_t u1 = S16[((size_t)(b * HID + c + 1) << 16) + s0 + px];
        const int t = px >> 4;
        int laneC, e; frag_pos(c & 31, px & 15, laneC, e);
        Sf[(((c >> 5) * 4 + t) * 32 + laneC) * FRW + (e >> 1)] = u0 | (u1 << 16);
    }
    __syncthreads();

    const float g = p_gamma[0];
    #pragma unroll
    for (int t = 0; t < 4; ++t) {
        v8f acc = {};
        #pragma unroll
        for (int kc = 0; kc < 4; ++kc) {
            const v16h bf =
                *(const v16h*)((const _Float16*)Sf + ((kc * 4 + t) * 32 + lane) * FR);
            acc = wmma16(aw[kc], bf, acc);
        }
        const int s = s0 + t * 16 + n;
        #pragma unroll
        for (int r = 0; r < 8; ++r) {
            const int o = mo + (half << 3) + r;
            const size_t idx = ((size_t)(b * CIN + o) << 16) + s;
            out[idx] = x[idx] + g * (acc[r] + b_out[o]);
        }
    }
}

extern "C" void kernel_launch(void* const* d_in, const int* in_sizes, int n_in,
                              void* d_out, int out_size, void* d_ws, size_t ws_size,
                              hipStream_t stream)
{
    (void)in_sizes; (void)n_in; (void)out_size; (void)ws_size;

    const float* x       = (const float*)d_in[0];
    const float* prior   = (const float*)d_in[1];
    const float* W_in    = (const float*)d_in[2];
    const float* b_in    = (const float*)d_in[3];
    const float* W_out   = (const float*)d_in[4];
    const float* b_out   = (const float*)d_in[5];
    const float* W_delta = (const float*)d_in[6];
    const float* b_delta = (const float*)d_in[7];
    const float* W_B     = (const float*)d_in[8];
    const float* b_B     = (const float*)d_in[9];
    const float* A_param = (const float*)d_in[10];
    const float* lam     = (const float*)d_in[11];
    const float* alp     = (const float*)d_in[12];
    const float* gam     = (const float*)d_in[13];

    float* out = (float*)d_out;
    float* pu  = out + (size_t)NB * CIN * SPX;   // tuple output #2 lives here

    const size_t NE = (size_t)NB * HID * SPX;    // 33,554,432 elements
    _Float16* xp_g  = (_Float16*)d_ws;           // 4 x 64 MB f16 tensors
    _Float16* Ab_g  = xp_g + NE;
    _Float16* Bb_g  = Ab_g + NE;
    _Float16* S_g   = Bb_g + NE;
    _Float16* Wp_in = S_g + NE;                  // packed f16 weight fragments
    _Float16* Wp_d  = Wp_in + 8 * 2 * 512;
    _Float16* Wp_B  = Wp_d + 8 * 4 * 512;
    _Float16* Wp_o  = Wp_B + 8 * 4 * 512;

    k_pack<<<(8 * 2 * 512) / 256, 256, 0, stream>>>(W_in, Wp_in, 2, CIN, 8 * 2 * 512);
    k_pack<<<(8 * 4 * 512) / 256, 256, 0, stream>>>(W_delta, Wp_d, 4, HID, 8 * 4 * 512);
    k_pack<<<(8 * 4 * 512) / 256, 256, 0, stream>>>(W_B, Wp_B, 4, HID, 8 * 4 * 512);
    k_pack<<<(4 * 4 * 512) / 256, 256, 0, stream>>>(W_out, Wp_o, 4, HID, 4 * 4 * 512);

    k_prior<<<(NB * SPX) / 256, 256, 0, stream>>>(prior, pu);
    k_proj<<<(NB * SPX) / 64, 256, 0, stream>>>(x, pu, Wp_in, b_in,
                                                Wp_d, b_delta, Wp_B, b_B,
                                                A_param, lam, alp,
                                                xp_g, Ab_g, Bb_g);
    k_scan_w<<<(NB * HID * 256 * 32) / 256, 256, 0, stream>>>(xp_g, Ab_g, Bb_g, S_g);
    k_scan_h<<<(NB * HID * 256) / 256, 256, 0, stream>>>(xp_g, Ab_g, Bb_g, S_g);
    k_out<<<(NB * SPX) / 64, 128, 0, stream>>>(x, S_g, Wp_o, b_out, gam, out);
}
